// GatedTitansMAC_46772193853521
// MI455X (gfx1250) — compile-verified, matrix operand order checked by reading, not run on hardware
//
#include <hip/hip_runtime.h>
#include <math.h>

// GatedTitansMAC for MI455X (gfx1250).
// All GEMMs on v_wmma_f32_16x16x32_bf16; LDS tiles fed by the Tensor Data
// Mover (tensor_load_to_lds, TENSORcnt) with double buffering so DMA of the
// next 128x32 tile overlaps WMMA on the current one.
// B=32768, H=1024, M=64; ~370 GFLOP vs ~0.5GB HBM -> matrix-pipe bound.

#define BB 32768
#define HH 1024
#define MM 64
#define TP 40                 // LDS tile pitch in bf16 elems (80B rows)
#define TILE_E (128 * TP)     // elems per LDS tile buffer
#define TILE_BYTES (TILE_E * 2)
#define LRc 0.01f
#define DECAYc 0.99f

typedef unsigned short u16;
typedef __attribute__((ext_vector_type(16))) __bf16 v16bf;
typedef __attribute__((ext_vector_type(8)))  float  v8f;
typedef __attribute__((ext_vector_type(4))) unsigned int u32x4;
typedef __attribute__((ext_vector_type(8))) int i32x8;
typedef __attribute__((ext_vector_type(4))) int i32x4;

#if defined(__has_builtin)
#if __has_builtin(__builtin_amdgcn_tensor_load_to_lds) && \
    __has_builtin(__builtin_amdgcn_s_wait_tensorcnt)
#define HAVE_TDM 1
#endif
#endif

struct U8x32 { uint4 lo, hi; };   // 32 bytes == v16bf

__device__ __forceinline__ u16 f2bf(float f) {
  unsigned u = __builtin_bit_cast(unsigned, f);
  u += 0x7FFFu + ((u >> 16) & 1u);              // round-to-nearest-even
  return (u16)(u >> 16);
}
__device__ __forceinline__ float bf2f(u16 h) {
  unsigned u = ((unsigned)h) << 16;
  return __builtin_bit_cast(float, u);
}
__device__ __forceinline__ int imin(int a, int b) { return a < b ? a : b; }

__device__ __forceinline__ v8f wmma_bf16(v16bf a, v16bf b, v8f c) {
  return __builtin_amdgcn_wmma_f32_16x16x32_bf16(false, a, false, b, (short)0, c,
                                                 false, false);
}

// 16x32 bf16 fragment from LDS tile (row-major [outer][k], pitch TP).
// ISA layout: lane L -> outer = L&15; elems 0..7 : k = 8*(L>>4)+0..7,
//                                   elems 8..15: k = 16+8*(L>>4)+0..7.
__device__ __forceinline__ v16bf frag_lds(const u16* tile, int rowBase, int lane) {
  int r  = rowBase + (lane & 15);
  int ko = (lane >> 4) * 8;
  const u16* p = tile + r * TP + ko;
  U8x32 d;
  d.lo = *(const uint4*)(p);
  d.hi = *(const uint4*)(p + 16);
  return __builtin_bit_cast(v16bf, d);
}

#if HAVE_TDM
// TDM: DMA one 128-row x 32-elem bf16 tile (64B rows) global->LDS.
// pad_interval=3 (16 DWORDs) + pad_amount=3 (4 DWORDs) -> 80B LDS pitch == TP.
// tensor_dim1 = validRows: rows past the matrix edge are written as zeros.
__device__ __forceinline__ void tdm_load_tile(unsigned ldsAddr, const u16* gsrc,
                                              int ld, int validRows) {
  unsigned long long ga = (unsigned long long)(size_t)gsrc;
  u32x4 g0;
  g0[0] = 1u;                                              // count=1 (valid)
  g0[1] = ldsAddr;                                         // lds_addr
  g0[2] = (unsigned)(ga & 0xFFFFFFFFu);                    // global_addr lo
  g0[3] = (unsigned)((ga >> 32) & 0x01FFFFFFu) | (2u << 30);  // addr hi | type=2
  i32x8 g1;
  g1[0] = (1 << 16) | (1 << 20) | (3 << 22) | (3 << 25);   // 2B elems, padding
  g1[1] = (ld & 0xFFFF) << 16;                             // tensor_dim0 lo16
  g1[2] = ((ld >> 16) & 0xFFFF) | ((validRows & 0xFFFF) << 16);
  g1[3] = ((validRows >> 16) & 0xFFFF) | (32 << 16);       // dim1 hi | tile_dim0=32
  g1[4] = 128;                                             // tile_dim1=128
  g1[5] = ld;                                              // dim0_stride lo32
  g1[6] = 0;
  g1[7] = 0;
  i32x4 z4 = {0, 0, 0, 0};
  i32x8 z8 = {0, 0, 0, 0, 0, 0, 0, 0};
  // 6-arg toolchain form: (group0, group1, group2, group3, group4/pad, cpol)
  __builtin_amdgcn_tensor_load_to_lds(g0, g1, z4, z4, z8, 0);
}
#endif

// ---------- core: C_tile(128x128) += A(MxK,row-major) * Bt(NxK,row-major)^T ----
// As/Bs are double-buffered (2*TILE_E elems each). lda==K and ldb==K for every
// call site, so a k-step tile starts at base + row0*ld + k0 with row stride ld.
__device__ __forceinline__ void gemm_nt_core(
    const u16* __restrict__ A, int lda, int aRow0, int aValid,
    const u16* __restrict__ Bt, int ldb, int bRow0, int bValid,
    int K, v8f (*acc)[4], u16* As, u16* Bs) {
  const int t = (int)threadIdx.x;
  const int lane = t & 31, wid = t >> 5;
  const int wm = (wid & 3) * 32, wn = (wid >> 2) * 64;
  __syncthreads();   // protect prologue DMA vs. prior reads of these buffers
#if HAVE_TDM
  const unsigned ldsA = (unsigned)(size_t)As;
  const unsigned ldsB = (unsigned)(size_t)Bs;
  if (wid == 0) {
    tdm_load_tile(ldsA, A + (size_t)aRow0 * lda, lda, aValid);
    tdm_load_tile(ldsB, Bt + (size_t)bRow0 * ldb, ldb, bValid);
  }
  int cur = 0;
  for (int k0 = 0; k0 < K; k0 += 32) {
    if (wid == 0) __builtin_amdgcn_s_wait_tensorcnt(0);
    __syncthreads();
    if (wid == 0 && k0 + 32 < K) {
      int nxt = cur ^ 1;
      tdm_load_tile(ldsA + nxt * TILE_BYTES,
                    A + (size_t)aRow0 * lda + k0 + 32, lda, aValid);
      tdm_load_tile(ldsB + nxt * TILE_BYTES,
                    Bt + (size_t)bRow0 * ldb + k0 + 32, ldb, bValid);
    }
    const u16* Ac = As + cur * TILE_E;
    const u16* Bc = Bs + cur * TILE_E;
    v16bf a0 = frag_lds(Ac, wm, lane);
    v16bf a1 = frag_lds(Ac, wm + 16, lane);
#pragma unroll
    for (int j = 0; j < 4; ++j) {
      v16bf b = frag_lds(Bc, wn + j * 16, lane);
      acc[0][j] = wmma_bf16(a0, b, acc[0][j]);
      acc[1][j] = wmma_bf16(a1, b, acc[1][j]);
    }
    cur ^= 1;
  }
#else
  const int frow = t >> 1, fseg = (t & 1) * 16;
  for (int k0 = 0; k0 < K; k0 += 32) {
    __syncthreads();
    {
      uint4 lo = make_uint4(0u, 0u, 0u, 0u), hi = lo;
      if (frow < aValid) {
        const uint4* p = (const uint4*)(A + (size_t)(aRow0 + frow) * lda + k0 + fseg);
        lo = p[0]; hi = p[1];
      }
      uint4* q = (uint4*)(As + frow * TP + fseg);
      q[0] = lo; q[1] = hi;
    }
    {
      uint4 lo = make_uint4(0u, 0u, 0u, 0u), hi = lo;
      if (frow < bValid) {
        const uint4* p = (const uint4*)(Bt + (size_t)(bRow0 + frow) * ldb + k0 + fseg);
        lo = p[0]; hi = p[1];
      }
      uint4* q = (uint4*)(Bs + frow * TP + fseg);
      q[0] = lo; q[1] = hi;
    }
    __syncthreads();
    v16bf a0 = frag_lds(As, wm, lane);
    v16bf a1 = frag_lds(As, wm + 16, lane);
#pragma unroll
    for (int j = 0; j < 4; ++j) {
      v16bf b = frag_lds(Bs, wn + j * 16, lane);
      acc[0][j] = wmma_bf16(a0, b, acc[0][j]);
      acc[1][j] = wmma_bf16(a1, b, acc[1][j]);
    }
  }
#endif
}

// ---------- core: C_tile(128x128) += Xa(B x M)^T * Xb(B x N) over batch rows ----
__device__ __forceinline__ void gemm_tt_core(
    const u16* __restrict__ Xa, int lda, int m0, int mValid,
    const u16* __restrict__ Xb, int ldb, int n0, int nValid,
    int kStart, int kLen, v8f (*acc)[4], u16* As, u16* Bs) {
  const int t = (int)threadIdx.x;
  const int lane = t & 31, wid = t >> 5;
  const int wm = (wid & 3) * 32, wn = (wid >> 2) * 64;
  const int kk = t & 31, mseg = (t >> 5) * 16;
  for (int k0 = kStart; k0 < kStart + kLen; k0 += 32) {
    __syncthreads();
    {
      const u16* p = Xa + (size_t)(k0 + kk) * lda + m0 + mseg;
#pragma unroll
      for (int i = 0; i < 16; ++i)
        As[(mseg + i) * TP + kk] = (mseg + i < mValid) ? p[i] : (u16)0;
    }
    {
      const u16* p = Xb + (size_t)(k0 + kk) * ldb + n0 + mseg;
#pragma unroll
      for (int i = 0; i < 16; ++i)
        Bs[(mseg + i) * TP + kk] = (mseg + i < nValid) ? p[i] : (u16)0;
    }
    __syncthreads();
    v16bf a0 = frag_lds(As, wm, lane);
    v16bf a1 = frag_lds(As, wm + 16, lane);
#pragma unroll
    for (int j = 0; j < 4; ++j) {
      v16bf b = frag_lds(Bs, wn + j * 16, lane);
      acc[0][j] = wmma_bf16(a0, b, acc[0][j]);
      acc[1][j] = wmma_bf16(a1, b, acc[1][j]);
    }
  }
}

#define ACC_DECL                                                         \
  v8f acc[2][4];                                                         \
  {                                                                      \
    const v8f vz = {0.f, 0.f, 0.f, 0.f, 0.f, 0.f, 0.f, 0.f};             \
    for (int i = 0; i < 2; ++i)                                          \
      for (int j = 0; j < 4; ++j) acc[i][j] = vz;                        \
  }

#define EPI_COORDS                                                       \
  int lane = (int)threadIdx.x & 31, wid = (int)threadIdx.x >> 5;         \
  int rB = (int)blockIdx.x * 128 + (wid & 3) * 32 + (lane >> 4) * 8;     \
  int cB = (int)blockIdx.y * 128 + (wid >> 2) * 64 + (lane & 15);

// ---------- generic GEMM: out = [relu]( A @ Bt^T + bias ) ----------
__global__ __launch_bounds__(256) void k_gemm(
    const u16* __restrict__ A, int lda, const u16* __restrict__ Bt, int ldb,
    const float* __restrict__ bias, u16* outBf, float* outF,
    int Mtot, int N, int K, int relu) {
  __shared__ __align__(128) u16 As[2 * TILE_E];
  __shared__ __align__(128) u16 Bs[2 * TILE_E];
  ACC_DECL;
  int aRow0 = (int)blockIdx.x * 128, bRow0 = (int)blockIdx.y * 128;
  gemm_nt_core(A, lda, aRow0, imin(128, Mtot - aRow0),
               Bt, ldb, bRow0, imin(128, N - bRow0), K, acc, As, Bs);
  EPI_COORDS;
#pragma unroll
  for (int i = 0; i < 2; ++i)
#pragma unroll
    for (int j = 0; j < 4; ++j)
#pragma unroll
      for (int r = 0; r < 8; ++r) {
        int row = rB + i * 16 + r, col = cB + j * 16;
        if (row < Mtot && col < N) {
          float v = acc[i][j][r];
          if (bias) v += bias[col];
          if (relu) v = v > 0.f ? v : 0.f;
          size_t idx = (size_t)row * N + col;
          if (outBf) outBf[idx] = f2bf(v);
          if (outF) outF[idx] = v;
        }
      }
}

// ---------- err = a @ W2 - v ; ssq[row] += sum(err^2) ----------
__global__ __launch_bounds__(256) void k_err(
    const u16* __restrict__ aBf, const u16* __restrict__ W2t,
    const float* __restrict__ vF, u16* errBf, float* ssq) {
  __shared__ __align__(128) u16 As[2 * TILE_E];
  __shared__ __align__(128) u16 Bs[2 * TILE_E];
  ACC_DECL;
  gemm_nt_core(aBf, MM, (int)blockIdx.x * 128, 128,
               W2t, MM, (int)blockIdx.y * 128, 128, MM, acc, As, Bs);
  EPI_COORDS;
#pragma unroll
  for (int i = 0; i < 2; ++i)
#pragma unroll
    for (int r = 0; r < 8; ++r) {
      int row = rB + i * 16 + r;
      float sq = 0.f;
#pragma unroll
      for (int j = 0; j < 4; ++j) {
        int col = cB + j * 16;
        float e = acc[i][j][r] - vF[(size_t)row * HH + col];
        errBf[(size_t)row * HH + col] = f2bf(e);
        sq += e * e;
      }
      atomicAdd(&ssq[row], sq);
    }
}

// ---------- da = (g_err @ W2^T) * (a > 0) ----------
__global__ __launch_bounds__(256) void k_da(
    const u16* __restrict__ gerrBf, const u16* __restrict__ W2b,
    const u16* __restrict__ aBf, u16* daBf) {
  __shared__ __align__(128) u16 As[2 * TILE_E];
  __shared__ __align__(128) u16 Bs[2 * TILE_E];
  ACC_DECL;
  gemm_nt_core(gerrBf, HH, (int)blockIdx.x * 128, 128,
               W2b, HH, 0, MM, HH, acc, As, Bs);
  EPI_COORDS;
#pragma unroll
  for (int i = 0; i < 2; ++i)
#pragma unroll
    for (int j = 0; j < 4; ++j)
#pragma unroll
      for (int r = 0; r < 8; ++r) {
        int row = rB + i * 16 + r, col = cB + j * 16;
        if (col < MM) {
          float v = acc[i][j][r];
          if (!(bf2f(aBf[(size_t)row * MM + col]) > 0.f)) v = 0.f;
          daBf[(size_t)row * MM + col] = f2bf(v);
        }
      }
}

// ---------- dW += Xa^T @ Xb  (split-K over batch, f32 atomics) ----------
__global__ __launch_bounds__(256) void k_dw(
    const u16* __restrict__ Xa, int lda, const u16* __restrict__ Xb, int ldb,
    float* C, int Mtot, int Ntot, int kChunk) {
  __shared__ __align__(128) u16 As[2 * TILE_E];
  __shared__ __align__(128) u16 Bs[2 * TILE_E];
  ACC_DECL;
  int m0 = (int)blockIdx.x * 128, n0 = (int)blockIdx.y * 128;
  gemm_tt_core(Xa, lda, m0, imin(128, Mtot - m0),
               Xb, ldb, n0, imin(128, Ntot - n0),
               (int)blockIdx.z * kChunk, kChunk, acc, As, Bs);
  EPI_COORDS;
#pragma unroll
  for (int i = 0; i < 2; ++i)
#pragma unroll
    for (int j = 0; j < 4; ++j)
#pragma unroll
      for (int r = 0; r < 8; ++r) {
        int row = rB + i * 16 + r, col = cB + j * 16;
        if (row < Mtot && col < Ntot)
          atomicAdd(&C[(size_t)row * Ntot + col], acc[i][j][r]);
      }
}

// ---------- out = hidden@WoTop^T + mem_ctx@WoBot^T + bo ----------
__global__ __launch_bounds__(256) void k_out(
    const u16* __restrict__ A1, const u16* __restrict__ Bt1,
    const u16* __restrict__ A2, const u16* __restrict__ Bt2,
    const float* __restrict__ bias, float* outF) {
  __shared__ __align__(128) u16 As[2 * TILE_E];
  __shared__ __align__(128) u16 Bs[2 * TILE_E];
  ACC_DECL;
  int m0 = (int)blockIdx.x * 128, n0 = (int)blockIdx.y * 128;
  gemm_nt_core(A1, HH, m0, 128, Bt1, HH, n0, 128, HH, acc, As, Bs);
  gemm_nt_core(A2, HH, m0, 128, Bt2, HH, n0, 128, HH, acc, As, Bs);
  EPI_COORDS;
#pragma unroll
  for (int i = 0; i < 2; ++i)
#pragma unroll
    for (int j = 0; j < 4; ++j)
#pragma unroll
      for (int r = 0; r < 8; ++r) {
        int row = rB + i * 16 + r, col = cB + j * 16;
        outF[(size_t)row * HH + col] = acc[i][j][r] + bias[col];
      }
}

// ---------- small helpers ----------
__global__ void k_cvt(const float* __restrict__ s, u16* d, int n) {
  int i = (int)blockIdx.x * 256 + (int)threadIdx.x;
  if (i < n) d[i] = f2bf(s[i]);
}
__global__ void k_zero(float* p, int n) {
  int i = (int)blockIdx.x * 256 + (int)threadIdx.x;
  if (i < n) p[i] = 0.f;
}
// dst (srcCols x srcRows) bf16 = transpose(src f32 rows [srcRow0, srcRow0+srcRows))
__global__ void k_transpose_bf(const float* __restrict__ src, int srcRow0,
                               int srcRows, int srcCols, u16* dst) {
  __shared__ float tile[32][33];
  int c0 = (int)blockIdx.x * 32, r0 = (int)blockIdx.y * 32;
  int tx = (int)threadIdx.x & 31, ty = (int)threadIdx.x >> 5;
  for (int yy = ty; yy < 32; yy += 8)
    tile[yy][tx] = src[(size_t)(srcRow0 + r0 + yy) * srcCols + c0 + tx];
  __syncthreads();
  for (int yy = ty; yy < 32; yy += 8)
    dst[(size_t)(c0 + yy) * srcRows + r0 + tx] = f2bf(tile[tx][yy]);
}
__global__ void k_hg(const float* __restrict__ hidden, const float* __restrict__ Wg,
                     float* hg) {
  int lane = (int)threadIdx.x & 31, wid = (int)threadIdx.x >> 5;
  int row = (int)blockIdx.x * 8 + wid;
  const float* hr = hidden + (size_t)row * HH;
  float s = 0.f;
  for (int c = lane; c < HH; c += 32) s += hr[c] * Wg[c];
  for (int off = 16; off > 0; off >>= 1) s += __shfl_xor(s, off, 32);
  if (lane == 0) hg[row] = s;
}
__global__ void k_gate(const float* __restrict__ hg, const float* __restrict__ ssq,
                       const float* __restrict__ bg, const float* __restrict__ wsS,
                       const int* __restrict__ step, float* gsc) {
  int r = (int)blockIdx.x * 256 + (int)threadIdx.x;
  float surprise = sqrtf(ssq[r] * (1.0f / (float)HH));
  float x = hg[r] + wsS[0] * surprise + bg[0];
  float g = 1.f / (1.f + expf(-x));
  g *= powf(DECAYc, (float)step[0]);
  gsc[r] = g * (1.0f / (float)BB);
}
__global__ void k_gerr(const u16* __restrict__ errBf, const float* __restrict__ gsc,
                       u16* out) {
  size_t i = (size_t)blockIdx.x * 256 + threadIdx.x;
  int row = (int)(i >> 10);
  out[i] = f2bf(bf2f(errBf[i]) * gsc[row]);
}
__global__ void k_wupd(const float* __restrict__ W, const float* __restrict__ dW,
                       float* out, int n) {
  int i = (int)blockIdx.x * 256 + (int)threadIdx.x;
  if (i < n) out[i] = W[i] - LRc * dW[i];
}

extern "C" void kernel_launch(void* const* d_in, const int* in_sizes, int n_in,
                              void* d_out, int out_size, void* d_ws, size_t ws_size,
                              hipStream_t stream) {
  const float* hidden = (const float*)d_in[0];
  const float* Wq = (const float*)d_in[1];
  const float* bq = (const float*)d_in[2];
  const float* Wk = (const float*)d_in[3];
  const float* bk = (const float*)d_in[4];
  const float* Wv = (const float*)d_in[5];
  const float* bv = (const float*)d_in[6];
  const float* W1 = (const float*)d_in[7];
  const float* W2 = (const float*)d_in[8];
  const float* Wg = (const float*)d_in[9];
  const float* bg = (const float*)d_in[10];
  const float* wsS = (const float*)d_in[11];
  const float* Wo = (const float*)d_in[12];
  const float* bo = (const float*)d_in[13];
  const int* step = (const int*)d_in[14];
  float* out = (float*)d_out;

  // workspace carve (~540 MB)
  char* wp = (char*)d_ws;
  auto carve = [&](size_t bytes) -> void* {
    void* p = (void*)wp;
    wp += (bytes + 255) & ~(size_t)255;
    return p;
  };
  const size_t NBH = (size_t)BB * HH;
  u16* hiddenBf = (u16*)carve(NBH * 2);
  u16* WqT = (u16*)carve((size_t)HH * HH * 2);
  u16* WkT = (u16*)carve((size_t)HH * HH * 2);
  u16* WvT = (u16*)carve((size_t)HH * HH * 2);
  u16* W1t = (u16*)carve((size_t)MM * HH * 2);   // 64 x 1024
  u16* W2t = (u16*)carve((size_t)HH * MM * 2);   // 1024 x 64
  u16* W2b = (u16*)carve((size_t)MM * HH * 2);   // 64 x 1024 row-major bf16
  u16* WoTT = (u16*)carve((size_t)HH * HH * 2);
  u16* WoBT = (u16*)carve((size_t)HH * HH * 2);
  u16* qBf = (u16*)carve(NBH * 2);
  u16* kBf = (u16*)carve(NBH * 2);
  float* vF = (float*)carve(NBH * 4);
  u16* aqBf = (u16*)carve((size_t)BB * MM * 2);
  u16* aBf = (u16*)carve((size_t)BB * MM * 2);
  u16* memBf = (u16*)carve(NBH * 2);
  u16* errBf = (u16*)carve(NBH * 2);
  u16* gerrBf = (u16*)carve(NBH * 2);
  u16* daBf = (u16*)carve((size_t)BB * MM * 2);
  float* ssq = (float*)carve((size_t)BB * 4);
  float* hg = (float*)carve((size_t)BB * 4);
  float* gsc = (float*)carve((size_t)BB * 4);
  float* dW1 = (float*)carve((size_t)HH * MM * 4);
  float* dW2 = (float*)carve((size_t)MM * HH * 4);

  const int nBH = BB * HH;
  // 0) precision prep: bf16 copies / transposes of weights and activations
  k_cvt<<<nBH / 256, 256, 0, stream>>>(hidden, hiddenBf, nBH);
  k_transpose_bf<<<dim3(32, 32), 256, 0, stream>>>(Wq, 0, HH, HH, WqT);
  k_transpose_bf<<<dim3(32, 32), 256, 0, stream>>>(Wk, 0, HH, HH, WkT);
  k_transpose_bf<<<dim3(32, 32), 256, 0, stream>>>(Wv, 0, HH, HH, WvT);
  k_transpose_bf<<<dim3(2, 32), 256, 0, stream>>>(W1, 0, HH, MM, W1t);   // W1^T
  k_transpose_bf<<<dim3(32, 2), 256, 0, stream>>>(W2, 0, MM, HH, W2t);   // W2^T
  k_cvt<<<(MM * HH) / 256, 256, 0, stream>>>(W2, W2b, MM * HH);
  k_transpose_bf<<<dim3(32, 32), 256, 0, stream>>>(Wo, 0, HH, HH, WoTT);
  k_transpose_bf<<<dim3(32, 32), 256, 0, stream>>>(Wo, HH, HH, HH, WoBT);
  k_zero<<<BB / 256, 256, 0, stream>>>(ssq, BB);
  k_zero<<<(HH * MM) / 256, 256, 0, stream>>>(dW1, HH * MM);
  k_zero<<<(MM * HH) / 256, 256, 0, stream>>>(dW2, MM * HH);

  // 1) q, k, v projections
  k_gemm<<<dim3(256, 8), 256, 0, stream>>>(hiddenBf, HH, WqT, HH, bq, qBf,
                                           (float*)nullptr, BB, HH, HH, 0);
  k_gemm<<<dim3(256, 8), 256, 0, stream>>>(hiddenBf, HH, WkT, HH, bk, kBf,
                                           (float*)nullptr, BB, HH, HH, 0);
  k_gemm<<<dim3(256, 8), 256, 0, stream>>>(hiddenBf, HH, WvT, HH, bv, (u16*)nullptr,
                                           vF, BB, HH, HH, 0);
  // 2) memory-MLP hidden layers: a_q = relu(q@W1), a = relu(k@W1)
  k_gemm<<<dim3(256, 1), 256, 0, stream>>>(qBf, HH, W1t, HH, (float*)nullptr, aqBf,
                                           (float*)nullptr, BB, MM, HH, 1);
  k_gemm<<<dim3(256, 1), 256, 0, stream>>>(kBf, HH, W1t, HH, (float*)nullptr, aBf,
                                           (float*)nullptr, BB, MM, HH, 1);
  // 3) mem_ctx = a_q @ W2
  k_gemm<<<dim3(256, 8), 256, 0, stream>>>(aqBf, MM, W2t, MM, (float*)nullptr, memBf,
                                           (float*)nullptr, BB, HH, MM, 0);
  // 4) err = a@W2 - v, surprise sumsq
  k_err<<<dim3(256, 8), 256, 0, stream>>>(aBf, W2t, vF, errBf, ssq);
  // 5) gate
  k_hg<<<BB / 8, 256, 0, stream>>>(hidden, Wg, hg);
  k_gate<<<BB / 256, 256, 0, stream>>>(hg, ssq, bg, wsS, step, gsc);
  // 6) g_err = gate * err / B
  k_gerr<<<nBH / 256, 256, 0, stream>>>(errBf, gsc, gerrBf);
  // 7) da = (g_err @ W2^T) * (h1 > 0)
  k_da<<<dim3(256, 1), 256, 0, stream>>>(gerrBf, W2b, aBf, daBf);
  // 8) dW1 = k^T @ da ; dW2 = a^T @ g_err (split-K over B, 64 chunks of 512)
  k_dw<<<dim3(8, 1, 64), 256, 0, stream>>>(kBf, HH, daBf, MM, dW1, HH, MM, BB / 64);
  k_dw<<<dim3(1, 8, 64), 256, 0, stream>>>(aBf, MM, gerrBf, HH, dW2, MM, HH, BB / 64);
  // 9) out = hidden@WoTop^T + mem_ctx@WoBot^T + bo
  k_out<<<dim3(256, 8), 256, 0, stream>>>(hiddenBf, WoTT, memBf, WoBT, bo, out);
  // 10) weight updates into d_out tail
  k_wupd<<<(HH * MM) / 256, 256, 0, stream>>>(W1, dW1, out + NBH, HH * MM);
  k_wupd<<<(MM * HH) / 256, 256, 0, stream>>>(W2, dW2, out + NBH + (size_t)HH * MM,
                                              MM * HH);
}
